// genMaskPatchNew_80788334837906
// MI455X (gfx1250) — matrix operand order repaired and checked by reading, not compiled, not run on hardware
//
#include <hip/hip_runtime.h>
#include <hip/hip_bf16.h>

typedef __attribute__((ext_vector_type(16))) _Float16 v16h;
typedef __attribute__((ext_vector_type(8)))  float    v8f;

// ---------------------------------------------------------------------------
// Kernel 0: init workspace argmax/argmin keys (ws is poisoned by harness).
// ws64[2b+0] = packed max key (init 0), ws64[2b+1] = packed min key (init ~0)
// ---------------------------------------------------------------------------
__global__ void init_ws_kernel(unsigned long long* __restrict__ ws64) {
    int i = threadIdx.x;
    if (i < 256) ws64[i] = (i & 1) ? ~0ull : 0ull;
}

// ---------------------------------------------------------------------------
// Kernel 1: one workgroup per image (256 threads = 8 wave32).
//   Phase A: sigmoid(x0-x1) per pixel (float4 loads) -> sliding 32-wide,
//            stride-2 row sums stored TRANSPOSED in LDS as f16: rs_t[x][row]
//   Phase B: vertical windowed sum as banded-ones WMMA:
//            pooled(16x16 tile) = A(16x64 ones-band, f16) x R(64x16 rs, f16)
//            2x v_wmma_f32_16x16x32_f16 accumulate; B fragments are single
//            ds_load_b32 each thanks to the transposed layout. Branchless
//            packed-u64 argmax/argmin reduction, one atomic pair per thread.
// ---------------------------------------------------------------------------
__global__ __launch_bounds__(256) void pool_argmax_kernel(
    const float* __restrict__ infeat, unsigned long long* __restrict__ ws64)
{
    __shared__ float    s_band[32][260];   // one 32-row sigmoid band (16B-aligned rows)
    __shared__ _Float16 rs_t[120][266];    // rs_t[x][row]; rows dword-aligned

    const int b    = blockIdx.x;
    const int tid  = threadIdx.x;
    const int lane = tid & 31;
    const int wid  = tid >> 5;

    const float* ch0 = infeat + (size_t)b * 2u * 65536u;
    const float* ch1 = ch0 + 65536;
    const float4* c0v = (const float4*)ch0;
    const float4* c1v = (const float4*)ch1;

    // ---------------- Phase A: sigmoid + horizontal window sums -------------
    const int rloc = tid >> 6;          // 0..3  (row within quad)
    const int cq   = tid & 63;          // float4 column index
    const int cb   = cq << 2;           // float column base
    for (int band = 0; band < 8; ++band) {
        __syncthreads();                // s_band free from previous band
        const int rowbase = band * 32;
        #pragma unroll
        for (int q = 0; q < 8; ++q) {
            const int row  = rowbase + q * 4 + rloc;
            const int vidx = row * 64 + cq;            // coalesced float4
            const float4 x0 = c0v[vidx];
            const float4 x1 = c1v[vidx];
            float4 s;
            s.x = 1.0f / (1.0f + __expf(x1.x - x0.x));
            s.y = 1.0f / (1.0f + __expf(x1.y - x0.y));
            s.z = 1.0f / (1.0f + __expf(x1.z - x0.z));
            s.w = 1.0f / (1.0f + __expf(x1.w - x0.w));
            *(float4*)&s_band[q * 4 + rloc][cb] = s;
        }
        if (band < 7) {
            __builtin_prefetch(ch0 + (rowbase + 32) * 256 + tid, 0, 0);
            __builtin_prefetch(ch1 + (rowbase + 32) * 256 + tid, 0, 0);
        }
        __syncthreads();
        // sliding 32-wide, stride-2 window sums: 8 threads per row
        const int r   = tid >> 3;
        const int seg = tid & 7;
        const int xs  = seg * 15;
        const int xe  = min(xs + 15, 113);
        float sum = 0.f;
        #pragma unroll
        for (int i = 0; i < 32; ++i) sum += s_band[r][2 * xs + i];
        rs_t[xs][rowbase + r] = (_Float16)sum;
        for (int x = xs + 1; x < xe; ++x) {
            sum += s_band[r][2 * x + 30] + s_band[r][2 * x + 31]
                 - s_band[r][2 * x - 2]  - s_band[r][2 * x - 1];
            rs_t[x][rowbase + r] = (_Float16)sum;
        }
    }
    __syncthreads();

    // ---------------- Phase B: vertical window sum via WMMA -----------------
    const int m = lane & 15;   // A-matrix row / B,N column / C column
    const int h = lane >> 4;   // half-wave selector

    // Banded-ones A matrices for K-chunks 0 (k=0..31) and 1 (k=32..63):
    // A[m][k] = 1 iff 2m <= k < 2m+32   (16-bit A 16x32 VGPR layout, ISA 7.12.2)
    v16h a0, a1;
    #pragma unroll
    for (int v = 0; v < 8; ++v) {
        const int kb = (v < 4) ? (2 * v + 8 * h) : (16 + 2 * (v - 4) + 8 * h);
        #pragma unroll
        for (int s = 0; s < 2; ++s) {
            const int kk = kb + s;
            a0[2 * v + s] = (_Float16)(((kk      >= 2 * m) && (kk      < 2 * m + 32)) ? 1.0f : 0.0f);
            a1[2 * v + s] = (_Float16)(((kk + 32 >= 2 * m) && (kk + 32 < 2 * m + 32)) ? 1.0f : 0.0f);
        }
    }

    unsigned long long maxk = 0ull, mink = ~0ull;
    const int y0 = wid * 16;          // wave wid owns output-row tile ty = wid

    for (int i = 0; i < 8; ++i) {     // x tiles
        const int x0   = i * 16;
        const int col  = x0 + m;                    // pooled x for this lane
        const int lcol = (col < 113) ? col : 112;   // clamp LDS column
        const unsigned int* cp = (const unsigned int*)(&rs_t[lcol][0]);
        // B (32x16 f16): VGPR v, half h -> K = (v<4 ? 8h + 2v : 16 + 8h + 2(v-4)) + {0,1}
        // transposed rs => rows K,K+1 are one aligned dword: single b32 load
        union { unsigned int u[8]; v16h v; } B0, B1;
        #pragma unroll
        for (int v = 0; v < 8; ++v) {
            const int K  = (v < 4) ? (8 * h + 2 * v) : (16 + 8 * h + 2 * (v - 4));
            const int r0 = 2 * y0 + K;                      // <= 254, even
            int       r1 = r0 + 32;                         // may exceed 254
            if (r1 > 254) r1 = 254;                         // clamp (discarded y)
            B0.u[v] = cp[r0 >> 1];
            B1.u[v] = cp[r1 >> 1];
        }
        v8f acc = {};
        acc = __builtin_amdgcn_wmma_f32_16x16x32_f16(false, a0, false, B0.v,
                                                     (short)0, acc, false, false);
        acc = __builtin_amdgcn_wmma_f32_16x16x32_f16(false, a1, false, B1.v,
                                                     (short)0, acc, false, false);
        // D layout: VGPR rr -> M = rr + 8h, N = lane&15. Branchless fold.
        const bool xok = (col < 113);
        #pragma unroll
        for (int rr = 0; rr < 8; ++rr) {
            const int  y  = y0 + rr + 8 * h;
            const bool ok = xok && (y < 113);
            const unsigned int fb  = __float_as_uint(acc[rr]);
            const unsigned int idx = (unsigned int)(y * 113 + col);
            const unsigned long long kmax =
                ok ? ((unsigned long long)fb << 32) | (unsigned int)(~idx) : 0ull;
            const unsigned long long kmin =
                ok ? ((unsigned long long)fb << 32) | idx : ~0ull;
            maxk = (kmax > maxk) ? kmax : maxk;
            mink = (kmin < mink) ? kmin : mink;
        }
    }
    atomicMax(&ws64[2 * b + 0], maxk);   // class 0 = argmax of pooled0
    atomicMin(&ws64[2 * b + 1], mink);   // class 1 = argmin of pooled0
}

// ---------------------------------------------------------------------------
// Kernel 2: decode winners -> oxy floats (d_out[0..1023]) + int boxes in ws
// oxy[b][c] = [[ox0,ox1],[oy0,oy1]]
// ---------------------------------------------------------------------------
__global__ void box_kernel(const unsigned long long* __restrict__ ws64,
                           float* __restrict__ out, int* __restrict__ boxes)
{
    const int b = threadIdx.x;
    if (b >= 128) return;
    const unsigned long long maxk = ws64[2 * b + 0];
    const unsigned long long mink = ws64[2 * b + 1];
    int ids[2];
    ids[0] = (int)(~(unsigned int)maxk);     // stored ~idx
    ids[1] = (int)((unsigned int)mink);      // stored idx
    #pragma unroll
    for (int c = 0; c < 2; ++c) {
        const int py  = ids[c] / 113, px = ids[c] % 113;
        const int ox0 = px * 2,       oy0 = py * 2;
        const int ox1 = min(255, ox0 + 31), oy1 = min(255, oy0 + 31);
        float* o = out + b * 8 + c * 4;
        o[0] = (float)ox0; o[1] = (float)ox1; o[2] = (float)oy0; o[3] = (float)oy1;
        int* bx = boxes + b * 8 + c * 4;
        bx[0] = ox0; bx[1] = ox1; bx[2] = oy0; bx[3] = oy1;
    }
}

// ---------------------------------------------------------------------------
// Kernel 3: mask = union of the two 32x32 boxes. float4 stores (32 MB total).
// Grid: 128 images * 64 row-quads; block 256 threads covers 4 rows.
// ---------------------------------------------------------------------------
__global__ __launch_bounds__(256) void mask_kernel(
    const int* __restrict__ boxes, float* __restrict__ mask)
{
    const int blk = blockIdx.x;                 // b*64 + rowquad
    const int b   = blk >> 6;
    const int hr  = ((blk & 63) << 2) + (threadIdx.x >> 6);
    const int w0  = (threadIdx.x & 63) << 2;
    const int* bx = boxes + b * 8;              // uniform per block -> scalar
    const bool yok0 = (hr >= bx[2]) & (hr <= bx[3]);
    const bool yok1 = (hr >= bx[6]) & (hr <= bx[7]);
    float4 o;
    float* po = &o.x;
    #pragma unroll
    for (int j = 0; j < 4; ++j) {
        const int w = w0 + j;
        const bool in0 = yok0 & (w >= bx[0]) & (w <= bx[1]);
        const bool in1 = yok1 & (w >= bx[4]) & (w <= bx[5]);
        po[j] = (in0 | in1) ? 1.0f : 0.0f;
    }
    *(float4*)&mask[(size_t)(((b << 8) + hr) << 8) + w0] = o;
}

// ---------------------------------------------------------------------------
extern "C" void kernel_launch(void* const* d_in, const int* in_sizes, int n_in,
                              void* d_out, int out_size, void* d_ws, size_t ws_size,
                              hipStream_t stream)
{
    const float* infeat = (const float*)d_in[0];           // [128,2,256,256]
    float* out = (float*)d_out;                            // 1024 oxy + 8.4M mask
    unsigned long long* ws64 = (unsigned long long*)d_ws;  // 256 u64 keys
    int* boxes = (int*)((char*)d_ws + 256 * sizeof(unsigned long long)); // 128*8 int

    init_ws_kernel<<<1, 256, 0, stream>>>(ws64);
    pool_argmax_kernel<<<128, 256, 0, stream>>>(infeat, ws64);
    box_kernel<<<1, 128, 0, stream>>>(ws64, out, boxes);
    mask_kernel<<<128 * 64, 256, 0, stream>>>(boxes, out + 1024);
}
